// GPT_14113262535208
// MI455X (gfx1250) — matrix operand order repaired and checked by reading, not compiled, not run on hardware
//
#include <hip/hip_runtime.h>
#include <hip/hip_bf16.h>
#include <stddef.h>
#include <stdint.h>

// ---------------------------------------------------------------------------
// GPT-2 small forward for gfx1250 (MI455X).
// bf16 WMMA for all GEMMs; Tensor Data Mover (TDM) feeds LDS tiles.
// ---------------------------------------------------------------------------

typedef __bf16 bf16;
typedef __attribute__((ext_vector_type(16))) __bf16 v16bf;
typedef __attribute__((ext_vector_type(8)))  __bf16 v8bf;
typedef __attribute__((ext_vector_type(8)))  float  v8f;
typedef __attribute__((ext_vector_type(4)))  unsigned int u32x4;
typedef __attribute__((ext_vector_type(8)))  int i32x8;
typedef __attribute__((ext_vector_type(4)))  int i32x4;

union V8F { v8f v; float f[8]; };

#if __has_builtin(__builtin_amdgcn_tensor_load_to_lds) && \
    __has_builtin(__builtin_amdgcn_s_wait_tensorcnt)
#define HAVE_TDM 1
#else
#define HAVE_TDM 0
#endif

__device__ __forceinline__ v8f wmma_bf16(v16bf a, v16bf b, v8f c) {
  return __builtin_amdgcn_wmma_f32_16x16x32_bf16(
      false, a, false, b, (short)0, c, false, false);
}

// A fragment (16x32 bf16): lanes 0-15 rows; elems 0..7 -> K=half*8+i,
// elems 8..15 -> K=16+half*8+(i-8). ld multiple of 8 elements.
__device__ __forceinline__ v16bf load_a_frag(const bf16* base, int ld,
                                             int row0, int k0, int lane) {
  const int r = lane & 15, hf = lane >> 4;
  const bf16* p = base + (size_t)(row0 + r) * ld + k0 + hf * 8;
  union { v16bf w; v8bf h[2]; } u;
  u.h[0] = *(const v8bf*)p;
  u.h[1] = *(const v8bf*)(p + 16);
  return u.w;
}

// B fragment (32x16 bf16) from K-transposed LDS tile [n][k]:
// lane column = col0+(lane&15), elems 0..7 -> K = k0+half*8+i, 8..15 -> K = k0+16+half*8+(i-8).
__device__ __forceinline__ v16bf load_bt_frag(const bf16* base, int ld,
                                              int col0, int k0, int lane) {
  const int c = lane & 15, hf = lane >> 4;
  const bf16* p = base + (size_t)(col0 + c) * ld + k0 + hf * 8;
  union { v16bf w; v8bf h[2]; } u;
  u.h[0] = *(const v8bf*)p;
  u.h[1] = *(const v8bf*)(p + 16);
  return u.w;
}

__device__ __forceinline__ float gelu_f(float x) {
  const float c = 0.7978845608028654f;  // sqrt(2/pi)
  float t = tanhf(c * (x + 0.044715f * x * x * x));
  return 0.5f * x * (1.0f + t);
}

// ---------------------------------------------------------------------------
// TDM: 2D bf16 tile DMA, global -> LDS, with LDS row padding.
// D# packing per CDNA5 ISA.
//   pad_icode: 0=2,1=4,2=8,3=16,4=32 DWORDs between pads
//   pad_acode: DWORDs of pad minus 1
// ---------------------------------------------------------------------------
__device__ __forceinline__ void tdm_load_2d(
    uint32_t lds_addr, const bf16* gsrc,
    uint32_t tile_d0, uint32_t tile_d1, uint32_t stride_d0,
    uint32_t pad_icode, uint32_t pad_acode) {
#if HAVE_TDM
  const uint64_t ga = (uint64_t)(uintptr_t)gsrc;
  u32x4 g0;
  g0[0] = 1u;                                    // count=1
  g0[1] = lds_addr;                              // LDS byte address
  g0[2] = (uint32_t)(ga & 0xffffffffu);          // global_addr[31:0]
  g0[3] = (uint32_t)((ga >> 32) & 0x01ffffffu) | (2u << 30);  // [56:32], type=2
  i32x8 g1;
  g1[0] = (int)((1u << 16) |                     // data_size = 2 bytes
                (1u << 20) |                     // pad_enable
                (pad_icode << 22) | (pad_acode << 25));
  g1[1] = (int)((tile_d0 & 0xffffu) << 16);      // tensor_dim0 = tile_d0
  g1[2] = (int)((tile_d0 >> 16) | ((tile_d1 & 0xffffu) << 16));  // tensor_dim1
  g1[3] = (int)((tile_d1 >> 16) | (tile_d0 << 16));              // tile_dim0
  g1[4] = (int)(tile_d1 & 0xffffu);              // tile_dim1 (tile_dim2=0)
  g1[5] = (int)stride_d0;                        // tensor_dim0_stride[31:0]
  g1[6] = 0;
  g1[7] = 0;
  i32x4 gz = {0, 0, 0, 0};
#if __clang_major__ >= 23
  i32x8 gz8 = {0, 0, 0, 0, 0, 0, 0, 0};
  __builtin_amdgcn_tensor_load_to_lds(g0, g1, gz, gz, gz8, 0);
#else
  __builtin_amdgcn_tensor_load_to_lds(g0, g1, gz, gz, 0);
#endif
#endif
}

__device__ __forceinline__ uint32_t lds_addr_of(const void* p) {
  return (uint32_t)(uintptr_t)p;
}

__device__ __forceinline__ void tdm_wait0() {
#if HAVE_TDM
  __builtin_amdgcn_s_wait_tensorcnt(0);
#endif
}

// ---------------------------------------------------------------------------
// Embedding: x[row,e] = wte[idx[row],e] + wpe[row%T,e]  (f32)
// ---------------------------------------------------------------------------
__global__ __launch_bounds__(256) void embed_kernel(
    const int* __restrict__ idx, const float* __restrict__ wte,
    const float* __restrict__ wpe, float* __restrict__ x) {
  const int row = blockIdx.x;
  const int t = row & 1023;
  const int tok = idx[row];
  const int e = threadIdx.x * 4;
  float4 a = *(const float4*)(wte + (size_t)tok * 1024 + e);
  float4 p = *(const float4*)(wpe + (size_t)t * 1024 + e);
  float4 o;
  o.x = a.x + p.x; o.y = a.y + p.y; o.z = a.z + p.z; o.w = a.w + p.w;
  *(float4*)(x + (size_t)row * 1024 + e) = o;
}

// ---------------------------------------------------------------------------
// Weight convert + transpose: src f32 [K,N] -> dst bf16 [N,K], per layer (z)
// ---------------------------------------------------------------------------
__global__ __launch_bounds__(256) void transpose_cvt(
    const float* __restrict__ src, bf16* __restrict__ dst, int K, int N) {
  __shared__ float tile[32][33];
  const size_t loff = (size_t)blockIdx.z * K * N;
  src += loff; dst += loff;
  const int k0 = blockIdx.y * 32, n0 = blockIdx.x * 32;
  const int tx = threadIdx.x & 31, ty = threadIdx.x >> 5;
#pragma unroll
  for (int j = 0; j < 4; ++j)
    tile[ty + j * 8][tx] = src[(size_t)(k0 + ty + j * 8) * N + n0 + tx];
  __syncthreads();
#pragma unroll
  for (int j = 0; j < 4; ++j)
    dst[(size_t)(n0 + ty + j * 8) * K + k0 + tx] = (bf16)tile[tx][ty + j * 8];
}

// ---------------------------------------------------------------------------
// LayerNorm over E=1024: in row = blockIdx.x*rmul+roff (f32), out bf16 or f32
// ---------------------------------------------------------------------------
template <typename OutTy>
__global__ __launch_bounds__(256) void ln_kernel(
    const float* __restrict__ x, const float* __restrict__ w,
    const float* __restrict__ bia, OutTy* __restrict__ out,
    int rmul, int roff) {
  __shared__ float s1[256], s2[256];
  const int tid = threadIdx.x;
  const size_t irow = (size_t)blockIdx.x * rmul + roff;
  const float* xr = x + irow * 1024;
  float4 v = *(const float4*)(xr + tid * 4);
  s1[tid] = v.x + v.y + v.z + v.w;
  s2[tid] = v.x * v.x + v.y * v.y + v.z * v.z + v.w * v.w;
  __syncthreads();
  for (int d = 128; d > 0; d >>= 1) {
    if (tid < d) { s1[tid] += s1[tid + d]; s2[tid] += s2[tid + d]; }
    __syncthreads();
  }
  const float mean = s1[0] * (1.0f / 1024.0f);
  const float var  = s2[0] * (1.0f / 1024.0f) - mean * mean;
  const float rs   = rsqrtf(var + 1e-5f);
  float4 wv = *(const float4*)(w + tid * 4);
  float4 bv = *(const float4*)(bia + tid * 4);
  OutTy* orow = out + (size_t)blockIdx.x * 1024 + tid * 4;
  orow[0] = (OutTy)((v.x - mean) * rs * wv.x + bv.x);
  orow[1] = (OutTy)((v.y - mean) * rs * wv.y + bv.y);
  orow[2] = (OutTy)((v.z - mean) * rs * wv.z + bv.z);
  orow[3] = (OutTy)((v.w - mean) * rs * wv.w + bv.w);
}

// ---------------------------------------------------------------------------
// bf16 WMMA GEMM with TDM-fed, double-buffered LDS tiles.
//   A  : bf16 [M,K] row-major        W t: bf16 [N,K] (pre-transposed)
//   EPI: 0 = bias, bf16 out; 1 = bias+gelu, bf16 out; 2 = bias+residual, f32
// Block tile 128x128, BK=32. 8 waves (2x4), wave tile 64x32 (4x2 WMMA accs).
// LDS rows padded to 40 elems (80 B) == TDM pad 4 DW every 16 DW.
// ---------------------------------------------------------------------------
#define G_LDS 40

template <int EPI>
__global__ __launch_bounds__(256) void gemm_bf16(
    const bf16* __restrict__ A, const bf16* __restrict__ Wt,
    const float* __restrict__ bias, const float* __restrict__ R,
    void* __restrict__ Cout, int M, int N, int K) {
  __shared__ __align__(16) bf16 sA[2][128 * G_LDS];
  __shared__ __align__(16) bf16 sW[2][128 * G_LDS];
  const int tid  = threadIdx.x;
  const int lane = tid & 31;
  const int wv   = tid >> 5;
  const int wm   = wv >> 2;   // 0..1
  const int wn   = wv & 3;    // 0..3
  const int m0   = blockIdx.y * 128;
  const int n0   = blockIdx.x * 128;
  const bf16* Abase = A + (size_t)m0 * K;
  const bf16* Wbase = Wt + (size_t)n0 * K;

  V8F acc[4][2];
#pragma unroll
  for (int a = 0; a < 4; ++a)
#pragma unroll
    for (int b2 = 0; b2 < 2; ++b2)
#pragma unroll
      for (int s = 0; s < 8; ++s) acc[a][b2].f[s] = 0.0f;

  const int niter = K >> 5;

#if HAVE_TDM
  // Prologue: DMA tile 0 into buffer 0 (wave0: A, wave1: W).
  if (wv == 0) tdm_load_2d(lds_addr_of(sA[0]), Abase, 32, 128, K, 3, 3);
  if (wv == 1) tdm_load_2d(lds_addr_of(sW[0]), Wbase, 32, 128, K, 3, 3);
  if (wv < 2) tdm_wait0();
  __syncthreads();
  for (int it = 0; it < niter; ++it) {
    const int cur = it & 1, nxt = cur ^ 1;
    if (it + 1 < niter) {
      const int kt = (it + 1) << 5;
      if (wv == 0) tdm_load_2d(lds_addr_of(sA[nxt]), Abase + kt, 32, 128, K, 3, 3);
      if (wv == 1) tdm_load_2d(lds_addr_of(sW[nxt]), Wbase + kt, 32, 128, K, 3, 3);
    }
    v16bf af[4], bfr[2];
#pragma unroll
    for (int a = 0; a < 4; ++a)
      af[a] = load_a_frag(sA[cur], G_LDS, wm * 64 + a * 16, 0, lane);
#pragma unroll
    for (int b2 = 0; b2 < 2; ++b2)
      bfr[b2] = load_bt_frag(sW[cur], G_LDS, wn * 32 + b2 * 16, 0, lane);
#pragma unroll
    for (int a = 0; a < 4; ++a)
#pragma unroll
      for (int b2 = 0; b2 < 2; ++b2)
        acc[a][b2].v = wmma_bf16(af[a], bfr[b2], acc[a][b2].v);
    __syncthreads();              // all reads of cur done
    if (it + 1 < niter) {
      if (wv < 2) tdm_wait0();    // next buffer landed
      __syncthreads();
    }
  }
#else
  // Fallback: synchronous cooperative copies, single buffer.
  for (int it = 0; it < niter; ++it) {
    const int kt = it << 5;
#pragma unroll
    for (int q = 0; q < 2; ++q) {
      const int id = tid + q * 256;          // 512 chunks of 8 bf16
      const int r = id >> 2, c0 = (id & 3) * 8;
      *(v8bf*)(sA[0] + r * G_LDS + c0) = *(const v8bf*)(Abase + (size_t)r * K + kt + c0);
      *(v8bf*)(sW[0] + r * G_LDS + c0) = *(const v8bf*)(Wbase + (size_t)r * K + kt + c0);
    }
    __syncthreads();
    v16bf af[4], bfr[2];
#pragma unroll
    for (int a = 0; a < 4; ++a)
      af[a] = load_a_frag(sA[0], G_LDS, wm * 64 + a * 16, 0, lane);
#pragma unroll
    for (int b2 = 0; b2 < 2; ++b2)
      bfr[b2] = load_bt_frag(sW[0], G_LDS, wn * 32 + b2 * 16, 0, lane);
#pragma unroll
    for (int a = 0; a < 4; ++a)
#pragma unroll
      for (int b2 = 0; b2 < 2; ++b2)
        acc[a][b2].v = wmma_bf16(af[a], bfr[b2], acc[a][b2].v);
    __syncthreads();
  }
#endif

  // Epilogue (template-specialized: straight-line)
  const int hf = lane >> 4, l15 = lane & 15;
#pragma unroll
  for (int a = 0; a < 4; ++a) {
#pragma unroll
    for (int b2 = 0; b2 < 2; ++b2) {
      const int col = n0 + wn * 32 + b2 * 16 + l15;
      const float bb = bias[col];
#pragma unroll
      for (int s = 0; s < 8; ++s) {
        const int row = m0 + wm * 64 + a * 16 + hf * 8 + s;
        float v = acc[a][b2].f[s] + bb;
        if (EPI == 2) {
          float* C = (float*)Cout;
          C[(size_t)row * N + col] = v + R[(size_t)row * N + col];
        } else {
          if (EPI == 1) v = gelu_f(v);
          bf16* C = (bf16*)Cout;
          C[(size_t)row * N + col] = (bf16)v;
        }
      }
    }
  }
}

// ---------------------------------------------------------------------------
// Flash attention, causal. Block = (b, h, 128-query tile); 8 waves, each a
// 16-row query strip. Q/K tiles via TDM, V transposed manually (overlaps DMA).
// qkv: bf16 [B*T, 3E]; out: bf16 [B*T, E]. Scores scaled 1/sqrt(D) in f32.
// ---------------------------------------------------------------------------
#define A_LD 72

__global__ __launch_bounds__(256) void attn_flash(
    const bf16* __restrict__ qkv, bf16* __restrict__ out) {
  const int Ttot = 1024, Eemb = 1024, QLD = 3072;
  __shared__ __align__(16) bf16 sQ [128 * A_LD];
  __shared__ __align__(16) bf16 sK [ 64 * A_LD];
  __shared__ __align__(16) bf16 sVt[ 64 * A_LD];  // [d][kcol]
  __shared__ __align__(16) bf16 sP [128 * A_LD];

  const int tid = threadIdx.x, lane = tid & 31, w = tid >> 5;
  const int hf = lane >> 4, l15 = lane & 15;
  const int bb = blockIdx.z, hh = blockIdx.y, qb = blockIdx.x * 128;
  const size_t rowbase = (size_t)bb * Ttot;
  const bf16* qbase = qkv + (rowbase + qb) * QLD + hh * 64;

#if HAVE_TDM
  if (w == 0) tdm_load_2d(lds_addr_of(sQ), qbase, 64, 128, QLD, 4, 3);
#else
#pragma unroll
  for (int q = 0; q < 4; ++q) {
    const int id = tid + q * 256;            // 1024 chunks of 8
    const int r = id >> 3, c0 = (id & 7) * 8;
    *(v8bf*)(sQ + r * A_LD + c0) = *(const v8bf*)(qbase + (size_t)r * QLD + c0);
  }
#endif

  V8F oacc[4];
  float m_[8], l_[8];
#pragma unroll
  for (int c = 0; c < 4; ++c)
#pragma unroll
    for (int s = 0; s < 8; ++s) oacc[c].f[s] = 0.0f;
#pragma unroll
  for (int s = 0; s < 8; ++s) { m_[s] = -1e30f; l_[s] = 0.0f; }

  const int nk = qb / 64 + 2;
  for (int j = 0; j < nk; ++j) {
    const int kb = j * 64;
    __syncthreads();  // previous sK/sVt consumers done
    const bf16* kbase = qkv + (rowbase + kb) * QLD + Eemb + hh * 64;
    const bf16* vbase = qkv + (rowbase + kb) * QLD + 2 * Eemb + hh * 64;
#if HAVE_TDM
    if (w == 0) tdm_load_2d(lds_addr_of(sK), kbase, 64, 64, QLD, 4, 3);
#else
#pragma unroll
    for (int q = 0; q < 2; ++q) {
      const int id = tid + q * 256;
      const int r = id >> 3, c0 = (id & 7) * 8;
      *(v8bf*)(sK + r * A_LD + c0) = *(const v8bf*)(kbase + (size_t)r * QLD + c0);
    }
#endif
    // V transpose (all threads) overlaps the K DMA
#pragma unroll
    for (int q = 0; q < 2; ++q) {
      const int id = tid + q * 256;
      const int r = id >> 3, c0 = (id & 7) * 8;
      v8bf vv = *(const v8bf*)(vbase + (size_t)r * QLD + c0);
#pragma unroll
      for (int e = 0; e < 8; ++e) sVt[(c0 + e) * A_LD + r] = vv[e];
    }
#if HAVE_TDM
    if (w == 0) tdm_wait0();   // Q (first iter) + K landed
#endif
    __syncthreads();

    // S = Q K^T  (per-wave 16x64)
    V8F sacc[4];
#pragma unroll
    for (int c = 0; c < 4; ++c)
#pragma unroll
      for (int s = 0; s < 8; ++s) sacc[c].f[s] = 0.0f;
#pragma unroll
    for (int k0 = 0; k0 < 64; k0 += 32) {
      v16bf aq = load_a_frag(sQ, A_LD, w * 16, k0, lane);
#pragma unroll
      for (int c = 0; c < 4; ++c) {
        v16bf bk = load_bt_frag(sK, A_LD, c * 16, k0, lane);
        sacc[c].v = wmma_bf16(aq, bk, sacc[c].v);
      }
    }

    // scale + causal mask + online softmax (row = hf*8+s of wave strip)
#pragma unroll
    for (int s = 0; s < 8; ++s) {
      const int qg = qb + w * 16 + hf * 8 + s;
      float mx = -1e30f;
#pragma unroll
      for (int c = 0; c < 4; ++c) {
        const int kg = kb + c * 16 + l15;
        float v = sacc[c].f[s] * 0.125f;  // 1/sqrt(64)
        if (kg > qg) v = -1e30f;
        sacc[c].f[s] = v;
        mx = fmaxf(mx, v);
      }
      for (int d = 1; d < 16; d <<= 1) mx = fmaxf(mx, __shfl_xor(mx, d, 32));
      const float nm  = fmaxf(m_[s], mx);
      const float fac = __expf(m_[s] - nm);
      m_[s] = nm;
      float sum = 0.0f;
#pragma unroll
      for (int c = 0; c < 4; ++c) {
        float p = __expf(sacc[c].f[s] - nm);
        sacc[c].f[s] = p;
        sum += p;
      }
      for (int d = 1; d < 16; d <<= 1) sum += __shfl_xor(sum, d, 32);
      l_[s] = l_[s] * fac + sum;
#pragma unroll
      for (int c = 0; c < 4; ++c) oacc[c].f[s] *= fac;
#pragma unroll
      for (int c = 0; c < 4; ++c)
        sP[(w * 16 + hf * 8 + s) * A_LD + c * 16 + l15] = (bf16)sacc[c].f[s];
    }
    __syncthreads();

    // O += P V
#pragma unroll
    for (int k0 = 0; k0 < 64; k0 += 32) {
      v16bf ap = load_a_frag(sP, A_LD, w * 16, k0, lane);
#pragma unroll
      for (int c = 0; c < 4; ++c) {
        v16bf bv = load_bt_frag(sVt, A_LD, c * 16, k0, lane);
        oacc[c].v = wmma_bf16(ap, bv, oacc[c].v);
      }
    }
  }

#pragma unroll
  for (int c = 0; c < 4; ++c)
#pragma unroll
    for (int s = 0; s < 8; ++s) {
      const int qg = qb + w * 16 + hf * 8 + s;
      out[(rowbase + qg) * Eemb + hh * 64 + c * 16 + l15] =
          (bf16)(oacc[c].f[s] / l_[s]);
    }
}

// ---------------------------------------------------------------------------
// Logits for the 8 last-position rows: out[b,v] = dot(xf8[b], wte[v])  (f32)
// ---------------------------------------------------------------------------
__global__ __launch_bounds__(256) void logits_kernel(
    const float* __restrict__ xf8, const float* __restrict__ wte,
    float* __restrict__ out, int Vv) {
  __shared__ float red[256];
  const int v = blockIdx.x;
  const int tid = threadIdx.x;
  float4 wv = *(const float4*)(wte + (size_t)v * 1024 + tid * 4);
  float acc[8];
#pragma unroll
  for (int b = 0; b < 8; ++b) {
    float4 xv = *(const float4*)(xf8 + (size_t)b * 1024 + tid * 4);
    acc[b] = wv.x * xv.x + wv.y * xv.y + wv.z * xv.z + wv.w * xv.w;
  }
#pragma unroll
  for (int b = 0; b < 8; ++b) {
    red[tid] = acc[b];
    __syncthreads();
    for (int d = 128; d > 0; d >>= 1) {
      if (tid < d) red[tid] += red[tid + d];
      __syncthreads();
    }
    if (tid == 0) out[(size_t)b * Vv + v] = red[0];
    __syncthreads();
  }
}

// ---------------------------------------------------------------------------
// Driver
// ---------------------------------------------------------------------------
extern "C" void kernel_launch(void* const* d_in, const int* in_sizes, int n_in,
                              void* d_out, int out_size, void* d_ws, size_t ws_size,
                              hipStream_t stream) {
  const int Vv = 50257, L = 12, E = 1024, T = 1024, B = 8;
  const int M = B * T;  // 8192

  const int*   idx    = (const int*)  d_in[0];
  const float* wte    = (const float*)d_in[1];
  const float* wpe    = (const float*)d_in[2];
  const float* ln1_w  = (const float*)d_in[3];
  const float* ln1_b  = (const float*)d_in[4];
  const float* attn_w = (const float*)d_in[5];
  const float* attn_b = (const float*)d_in[6];
  const float* proj_w = (const float*)d_in[7];
  const float* proj_b = (const float*)d_in[8];
  const float* ln2_w  = (const float*)d_in[9];
  const float* ln2_b  = (const float*)d_in[10];
  const float* fc_w   = (const float*)d_in[11];
  const float* fc_b   = (const float*)d_in[12];
  const float* fc2_w  = (const float*)d_in[13];
  const float* fc2_b  = (const float*)d_in[14];
  const float* lnf_w  = (const float*)d_in[15];
  const float* lnf_b  = (const float*)d_in[16];

  // Workspace carve (byte-based, 256B aligned)
  char* base = (char*)d_ws;
  size_t off = 0;
  auto carve = [&](size_t bytes) -> void* {
    void* p = base + off;
    off += (bytes + 255) & ~(size_t)255;
    return p;
  };
  float* x    = (float*)carve((size_t)M * E * 4);
  bf16*  h    = (bf16*) carve((size_t)M * E * 2);
  bf16*  qkv  = (bf16*) carve((size_t)M * 3 * E * 2);
  bf16*  yA   = (bf16*) carve((size_t)M * E * 2);
  bf16*  fcb  = (bf16*) carve((size_t)M * 4 * E * 2);
  float* xf8  = (float*)carve((size_t)B * E * 4);
  bf16*  attn_wt = (bf16*)carve((size_t)L * 3 * E * E * 2);
  bf16*  proj_wt = (bf16*)carve((size_t)L * E * E * 2);
  bf16*  fc_wt   = (bf16*)carve((size_t)L * 4 * E * E * 2);
  bf16*  fc2_wt  = (bf16*)carve((size_t)L * E * 4 * E * 2);
  (void)ws_size; (void)in_sizes; (void)n_in; (void)out_size;

  // Weight convert + transpose: [K,N] f32 -> [N,K] bf16 (halves GEMM traffic)
  {
    dim3 ga(3 * E / 32, E / 32, L);
    transpose_cvt<<<ga, 256, 0, stream>>>(attn_w, attn_wt, E, 3 * E);
    dim3 gp(E / 32, E / 32, L);
    transpose_cvt<<<gp, 256, 0, stream>>>(proj_w, proj_wt, E, E);
    dim3 gf(4 * E / 32, E / 32, L);
    transpose_cvt<<<gf, 256, 0, stream>>>(fc_w, fc_wt, E, 4 * E);
    dim3 g2(E / 32, 4 * E / 32, L);
    transpose_cvt<<<g2, 256, 0, stream>>>(fc2_w, fc2_wt, 4 * E, E);
  }

  embed_kernel<<<M, 256, 0, stream>>>(idx, wte, wpe, x);

  for (int i = 0; i < L; ++i) {
    ln_kernel<bf16><<<M, 256, 0, stream>>>(x, ln1_w + i * E, ln1_b + i * E, h, 1, 0);

    dim3 g1(3 * E / 128, M / 128);
    gemm_bf16<0><<<g1, 256, 0, stream>>>(h, attn_wt + (size_t)i * 3 * E * E,
                                         attn_b + (size_t)i * 3 * E, nullptr,
                                         qkv, M, 3 * E, E);

    dim3 ga(T / 128, 16, B);
    attn_flash<<<ga, 256, 0, stream>>>(qkv, yA);

    dim3 g2(E / 128, M / 128);
    gemm_bf16<2><<<g2, 256, 0, stream>>>(yA, proj_wt + (size_t)i * E * E,
                                         proj_b + (size_t)i * E, x, x, M, E, E);

    ln_kernel<bf16><<<M, 256, 0, stream>>>(x, ln2_w + i * E, ln2_b + i * E, h, 1, 0);

    dim3 g3(4 * E / 128, M / 128);
    gemm_bf16<1><<<g3, 256, 0, stream>>>(h, fc_wt + (size_t)i * 4 * E * E,
                                         fc_b + (size_t)i * 4 * E, nullptr,
                                         fcb, M, 4 * E, E);

    dim3 g4(E / 128, M / 128);
    gemm_bf16<2><<<g4, 256, 0, stream>>>(fcb, fc2_wt + (size_t)i * E * 4 * E,
                                         fc2_b + (size_t)i * E, x, x, M, E, 4 * E);
  }

  ln_kernel<float><<<B, 256, 0, stream>>>(x, lnf_w, lnf_b, xf8, T, T - 1);
  logits_kernel<<<Vv, 256, 0, stream>>>(xf8, wte, (float*)d_out, Vv);
}
